// DagLSTMACC_78116865180273
// MI455X (gfx1250) — matrix operand order repaired and checked by reading, not compiled, not run on hardware
//
#include <hip/hip_runtime.h>
#include <hip/hip_bf16.h>

#define NNODES 65536
#define NLAYERS 8
#define NEDGES 8192
#define SDIM 128
#define CHUNK 8192
#define NTYPES 9   // T+1 edge types (incl. zero type)
#define VOCAB 10000

typedef __attribute__((ext_vector_type(16))) __bf16 v16bf;
typedef __attribute__((ext_vector_type(8)))  __bf16 v8bf;
typedef __attribute__((ext_vector_type(8)))  float  v8f;
typedef __attribute__((ext_vector_type(4))) unsigned int u32x4;
typedef __attribute__((ext_vector_type(8))) int i32x8;
typedef __attribute__((ext_vector_type(4))) int i32x4;

#if defined(__HIP_DEVICE_COMPILE__) && __has_builtin(__builtin_amdgcn_tensor_load_to_lds)
#define USE_TDM 1
#else
#define USE_TDM 0
#endif

__device__ __forceinline__ unsigned short f32_to_bf16_bits(float f) {
  unsigned u = __float_as_uint(f);
  u += 0x7fffu + ((u >> 16) & 1u);           // round-to-nearest-even
  return (unsigned short)(u >> 16);
}

__device__ __forceinline__ float sigmoidf_fast(float x) {
  return 1.0f / (1.0f + __expf(-x));
}

// A-matrix 16x32 bf16 fragment (ISA 7.12.2): lane&15 = row M, lane>>4 selects
// K-halves {0..7,16..23} vs {8..15,24..31}. tile_k points at the k-step base of
// a row-major (stride SDIM) bf16 tile.
__device__ __forceinline__ v16bf load_a_frag(const unsigned short* tile_k, int lane) {
  int row = lane & 15;
  int k0  = (lane >> 4) << 3;                // 0 or 8
  const v8bf lo = *(const v8bf*)(tile_k + row * SDIM + k0);
  const v8bf hi = *(const v8bf*)(tile_k + row * SDIM + k0 + 16);
  return __builtin_shufflevector(lo, hi, 0,1,2,3,4,5,6,7,8,9,10,11,12,13,14,15);
}

// B-matrix 32x16 bf16 fragment: lane&15 = column N, lane>>4 selects
// K-halves {0..7,16..23} vs {8..15,24..31} (same order as A).
__device__ __forceinline__ v16bf load_b_frag(const unsigned short* B, int krow0,
                                             int ncol0, int lane) {
  int n  = ncol0 + (lane & 15);
  int kb = krow0 + ((lane >> 4) << 3);
  v16bf r;
#pragma unroll
  for (int s = 0; s < 8; ++s) {
    r[s]     = __builtin_bit_cast(__bf16, B[(size_t)(kb + s) * SDIM + n]);
    r[s + 8] = __builtin_bit_cast(__bf16, B[(size_t)(kb + 16 + s) * SDIM + n]);
  }
  return r;
}

// ---------------------------------------------------------------- utilities
__global__ void k_zero(float* p, int n) {
  int i = blockIdx.x * blockDim.x + threadIdx.x;
  if (i < n) p[i] = 0.0f;
}

__global__ void k_copy_out(const float* src, float* dst, int n) {
  int i = blockIdx.x * blockDim.x + threadIdx.x;
  if (i < n) dst[i] = src[i];
}

__global__ __launch_bounds__(256) void k_convert_weights(
    const float* Wi, const float* Wo, const float* Wc, const float* Wf,
    const float* Ui, const float* Uo, const float* Uc, const float* Uf,
    unsigned short* Wb, unsigned short* Ub) {
  int idx = blockIdx.x * blockDim.x + threadIdx.x;
  const int WTOT = 4 * SDIM * SDIM;                 // 65536
  const int UPER = NTYPES * SDIM * SDIM;            // 147456
  if (idx < WTOT) {
    int g = idx >> 14, off = idx & (SDIM * SDIM - 1);
    const float* W = (g == 0) ? Wi : (g == 1) ? Wo : (g == 2) ? Wc : Wf;
    Wb[idx] = f32_to_bf16_bits(W[off]);
  } else {
    int j = idx - WTOT;
    if (j < 4 * UPER) {
      int g = j / UPER, off = j % UPER;
      const float* U = (g == 0) ? Ui : (g == 1) ? Uo : (g == 2) ? Uc : Uf;
      Ub[j] = f32_to_bf16_bits(U[off]);
    }
  }
}

// ------------------------------------------------- per-layer chunk WX GEMMs
// One block = one 16-row tile of the layer chunk. 8 waves: wave = (gate<<1)|half.
// A-tile staging: TDM gather-mode descriptor fetches the 16 indexed embedding
// rows (node_ids < 2^16 -> 16-bit row indices) straight into LDS, tracked with
// TENSORcnt; fallback is a lane-level global gather. Then f32->bf16 in LDS and
// 4 gate GEMMs via WMMA bf16 with B operands streamed from L2-resident weights.
__global__ __launch_bounds__(256) void k_wx_chunk(
    const float* __restrict__ emb, const int* __restrict__ node_ids,
    const unsigned short* __restrict__ Wb, float* __restrict__ wx,
    int chunk_base) {
  __shared__ __align__(32) unsigned short As[16 * SDIM];
  int tid = threadIdx.x;
#if USE_TDM
  __shared__ __align__(16) float Ef[16 * SDIM];
  if (tid < 32) {                          // wave 0 issues the gather DMA
    int nb = chunk_base + blockIdx.x * 16;
    unsigned lds_off = (unsigned)(uintptr_t)(void*)Ef;
    unsigned long long ga = (unsigned long long)(uintptr_t)emb;
    u32x4 g0;
    g0[0] = 1u | (1u << 31);               // count=1, gather_mode=1, 16-bit idx
    g0[1] = lds_off;                       // lds_addr
    g0[2] = (unsigned)(ga & 0xffffffffu);  // global_addr lo
    g0[3] = (unsigned)(ga >> 32) | (2u << 30);  // global_addr hi | type=2
    i32x8 g1;
    g1[0] = (2 << 16);                     // data_size = 4B
    g1[1] = (SDIM & 0xffff) << 16;         // tensor_dim0 lo16
    g1[2] = (VOCAB & 0xffff) << 16;        // tensor_dim0 hi16 | tensor_dim1 lo16
    g1[3] = (SDIM & 0xffff) << 16;         // tensor_dim1 hi16 | tile_dim0=128
    g1[4] = 16;                            // tile_dim1 = 16 valid indices
    g1[5] = SDIM;                          // tensor_dim0_stride lo32
    g1[6] = 0;
    g1[7] = 0;
    i32x4 g2, g3;
#pragma unroll
    for (int j = 0; j < 4; ++j) {
      g2[j] = (node_ids[nb + 2 * j] & 0xffff) |
              ((node_ids[nb + 2 * j + 1] & 0xffff) << 16);
      g3[j] = (node_ids[nb + 8 + 2 * j] & 0xffff) |
              ((node_ids[nb + 8 + 2 * j + 1] & 0xffff) << 16);
    }
#if __clang_major__ >= 23
    i32x8 zg = {};
    __builtin_amdgcn_tensor_load_to_lds(g0, g1, g2, g3, zg, 0);
#else
    __builtin_amdgcn_tensor_load_to_lds(g0, g1, g2, g3, 0);
#endif
    __builtin_amdgcn_s_wait_tensorcnt(0);
  }
  __syncthreads();
  {  // f32 LDS -> bf16 LDS, 8 elements per thread
    int base = tid * 8;
#pragma unroll
    for (int j = 0; j < 8; ++j) As[base + j] = f32_to_bf16_bits(Ef[base + j]);
  }
#else
  {
    int row = tid >> 4, seg = tid & 15;
    int node = chunk_base + blockIdx.x * 16 + row;       // < NNODES always
    int gid  = node_ids[node];
    const float* src = emb + (size_t)gid * SDIM + seg * 8;
    float4 a0 = *(const float4*)(src);
    float4 a1 = *(const float4*)(src + 4);
    unsigned short* dst = As + row * SDIM + seg * 8;
    dst[0] = f32_to_bf16_bits(a0.x); dst[1] = f32_to_bf16_bits(a0.y);
    dst[2] = f32_to_bf16_bits(a0.z); dst[3] = f32_to_bf16_bits(a0.w);
    dst[4] = f32_to_bf16_bits(a1.x); dst[5] = f32_to_bf16_bits(a1.y);
    dst[6] = f32_to_bf16_bits(a1.z); dst[7] = f32_to_bf16_bits(a1.w);
  }
#endif
  __syncthreads();

  int lane = tid & 31, wave = tid >> 5;
  int gate = wave >> 1, half = wave & 1;
  const unsigned short* W = Wb + (size_t)gate * (SDIM * SDIM);

  v8f acc[4] = {};
#pragma unroll
  for (int kk = 0; kk < 4; ++kk) {
    v16bf a = load_a_frag(As + kk * 32, lane);
#pragma unroll
    for (int nt = 0; nt < 4; ++nt) {
      v16bf b = load_b_frag(W, kk * 32, half * 64 + nt * 16, lane);
      acc[nt] = __builtin_amdgcn_wmma_f32_16x16x32_bf16(
          false, a, false, b, (short)0, acc[nt], false, false);
    }
  }

  int col = lane & 15, rbase = (lane >> 4) * 8;
  float* out = wx + (size_t)gate * CHUNK * SDIM + (size_t)(blockIdx.x * 16) * SDIM;
#pragma unroll
  for (int nt = 0; nt < 4; ++nt) {
    int ncol = half * 64 + nt * 16 + col;
#pragma unroll
    for (int v = 0; v < 8; ++v)
      out[(v + rbase) * SDIM + ncol] = acc[nt][v];
  }
}

// ---------------------------------------------- gather child state per edge
__global__ void k_gather(const float* __restrict__ reprs,
                         const float* __restrict__ mem,
                         const int* __restrict__ src_l,
                         unsigned short* __restrict__ srep,
                         float* __restrict__ mem_src) {
  int idx = blockIdx.x * blockDim.x + threadIdx.x;     // NEDGES*SDIM
  int e = idx >> 7, c = idx & (SDIM - 1);
  int s = src_l[e];                                    // may be zero node N
  size_t g = (size_t)s * SDIM + c;
  srep[idx]    = f32_to_bf16_bits(reprs[g]);
  mem_src[idx] = mem[g];
}

// ------------------------------------------- edge messages + segment reduce
// One block = one 16-edge tile. Loop over edge types present (bitmask),
// D = srep_tile @ U_gate[t] via WMMA, scatter matching rows with f32 atomics.
// Forget gate fused: sigmoid(WfX[tgt]+mf+b_f) * mem[src] into acc[3].
__global__ __launch_bounds__(256) void k_edges(
    const unsigned short* __restrict__ srep, const float* __restrict__ mem_src,
    const unsigned short* __restrict__ Ub, const float* __restrict__ wxf,
    const int* __restrict__ tgt_l, const int* __restrict__ et_l,
    const float* __restrict__ bf_p, float* __restrict__ acc, int chunk_base) {
  __shared__ int s_et[16];
  __shared__ int s_tl[16];
  __shared__ unsigned s_tm[NTYPES];
  int tid = threadIdx.x;
  int tile = blockIdx.x;
  if (tid < NTYPES) s_tm[tid] = 0u;
  if (tid < 16) {
    int e = tile * 16 + tid;
    s_et[tid] = et_l[e];
    s_tl[tid] = tgt_l[e] - chunk_base;                 // targets live in chunk
  }
  __syncthreads();
  if (tid < 16) atomicOr(&s_tm[s_et[tid]], 1u << tid);
  __syncthreads();

  int lane = tid & 31, wave = tid >> 5;
  int gate = wave >> 1, half = wave & 1;
  float bf = bf_p[0];

  // Preload the whole A tile (16 edges x 128 K) as 4 fragments, reused per type.
  const unsigned short* abase = srep + (size_t)(tile * 16) * SDIM;
  v16bf afrag[4];
#pragma unroll
  for (int kk = 0; kk < 4; ++kk) afrag[kk] = load_a_frag(abase + kk * 32, lane);

  int colw = lane & 15, rbase = (lane >> 4) * 8;
  const size_t P = (size_t)CHUNK * SDIM;

  for (int t = 0; t < NTYPES; ++t) {
    if (s_tm[t] == 0u) continue;
    const unsigned short* U = Ub + (size_t)(gate * NTYPES + t) * (SDIM * SDIM);
    __builtin_prefetch(U + lane * 512, 0, 1);          // global_prefetch_b8
    v8f d[4] = {};
#pragma unroll
    for (int kk = 0; kk < 4; ++kk) {
#pragma unroll
      for (int nt = 0; nt < 4; ++nt) {
        v16bf b = load_b_frag(U, kk * 32, half * 64 + nt * 16, lane);
        d[nt] = __builtin_amdgcn_wmma_f32_16x16x32_bf16(
            false, afrag[kk], false, b, (short)0, d[nt], false, false);
      }
    }
#pragma unroll
    for (int nt = 0; nt < 4; ++nt) {
      int ncol = half * 64 + nt * 16 + colw;
#pragma unroll
      for (int v = 0; v < 8; ++v) {
        int r = v + rbase;
        if (s_et[r] != t) continue;
        int tl = s_tl[r];
        float val = d[nt][v];
        if (gate < 3) {
          atomicAdd(&acc[(size_t)gate * P + (size_t)tl * SDIM + ncol], val);
        } else {
          float pre = wxf[(size_t)tl * SDIM + ncol] + val + bf;
          float fg  = sigmoidf_fast(pre);
          atomicAdd(&acc[3 * P + (size_t)tl * SDIM + ncol],
                    fg * mem_src[(size_t)(tile * 16 + r) * SDIM + ncol]);
        }
      }
    }
  }
}

// ------------------------------------------------------- LSTM node update
__global__ void k_node_update(const float* __restrict__ wx,
                              const float* __restrict__ acc,
                              const float* __restrict__ bi,
                              const float* __restrict__ bo,
                              const float* __restrict__ bc,
                              float* __restrict__ reprs, float* __restrict__ mem,
                              int chunk_base) {
  int idx = blockIdx.x * blockDim.x + threadIdx.x;     // CHUNK*SDIM
  int n = idx >> 7, c = idx & (SDIM - 1);
  const size_t P = (size_t)CHUNK * SDIM;
  float i_ = sigmoidf_fast(wx[idx]           + acc[idx]           + bi[0]);
  float o_ = sigmoidf_fast(wx[P + idx]       + acc[P + idx]       + bo[0]);
  float cg = tanhf       (wx[2 * P + idx]    + acc[2 * P + idx]   + bc[0]);
  float par = i_ * cg + acc[3 * P + idx];
  size_t g = (size_t)(chunk_base + n) * SDIM + c;
  float m = mem[g] + par;
  mem[g] = m;
  reprs[g] += o_ * tanhf(m);
}

// --------------------------------------------------------------- launcher
extern "C" void kernel_launch(void* const* d_in, const int* in_sizes, int n_in,
                              void* d_out, int out_size, void* d_ws, size_t ws_size,
                              hipStream_t stream) {
  (void)in_sizes; (void)n_in; (void)out_size;
  const float* emb = (const float*)d_in[0];
  const float* Wi  = (const float*)d_in[1];
  const float* Wo  = (const float*)d_in[2];
  const float* Wc  = (const float*)d_in[3];
  const float* Wf  = (const float*)d_in[4];
  const float* Ui  = (const float*)d_in[5];
  const float* Uo  = (const float*)d_in[6];
  const float* Uc  = (const float*)d_in[7];
  const float* Uf  = (const float*)d_in[8];
  const float* b_i = (const float*)d_in[9];
  const float* b_o = (const float*)d_in[10];
  const float* b_c = (const float*)d_in[11];
  const float* b_f = (const float*)d_in[12];
  // d_in[13] = upd_mask (implied by chunk structure, unused)
  const int* node_ids = (const int*)d_in[14];
  const int* targets  = (const int*)d_in[15];
  const int* sources  = (const int*)d_in[16];
  const int* etypes   = (const int*)d_in[17];

  char* ws = (char*)d_ws;
  size_t off = 0;
  auto walloc = [&](size_t bytes) {
    char* p = ws + off;
    off += (bytes + 255) & ~(size_t)255;
    return (void*)p;
  };
  unsigned short* Wb = (unsigned short*)walloc((size_t)4 * SDIM * SDIM * 2);
  unsigned short* Ub = (unsigned short*)walloc((size_t)4 * NTYPES * SDIM * SDIM * 2);
  float* reprs = (float*)walloc((size_t)(NNODES + 1) * SDIM * 4);
  float* mem   = (float*)walloc((size_t)(NNODES + 1) * SDIM * 4);
  float* wx    = (float*)walloc((size_t)4 * CHUNK * SDIM * 4);
  float* acc   = (float*)walloc((size_t)4 * CHUNK * SDIM * 4);
  unsigned short* srep = (unsigned short*)walloc((size_t)NEDGES * SDIM * 2);
  float* mem_src = (float*)walloc((size_t)NEDGES * SDIM * 4);
  if (off > ws_size) return;

  {  // weights fp32 -> bf16
    int tot = 4 * SDIM * SDIM + 4 * NTYPES * SDIM * SDIM;
    k_convert_weights<<<(tot + 255) / 256, 256, 0, stream>>>(
        Wi, Wo, Wc, Wf, Ui, Uo, Uc, Uf, Wb, Ub);
  }
  {  // zero state (incl. zero-node row N)
    int n = (NNODES + 1) * SDIM;
    k_zero<<<(n + 255) / 256, 256, 0, stream>>>(reprs, n);
    k_zero<<<(n + 255) / 256, 256, 0, stream>>>(mem, n);
  }

  for (int l = 0; l < NLAYERS; ++l) {
    int cb = l * CHUNK;
    {
      int n = 4 * CHUNK * SDIM;
      k_zero<<<(n + 255) / 256, 256, 0, stream>>>(acc, n);
    }
    k_wx_chunk<<<CHUNK / 16, 256, 0, stream>>>(emb, node_ids, Wb, wx, cb);
    {
      int n = NEDGES * SDIM;
      k_gather<<<(n + 255) / 256, 256, 0, stream>>>(
          reprs, mem, sources + l * NEDGES, srep, mem_src);
    }
    k_edges<<<NEDGES / 16, 256, 0, stream>>>(
        srep, mem_src, Ub, wx + (size_t)3 * CHUNK * SDIM,
        targets + l * NEDGES, etypes + l * NEDGES, b_f, acc, cb);
    k_node_update<<<(CHUNK * SDIM + 255) / 256, 256, 0, stream>>>(
        wx, acc, b_i, b_o, b_c, reprs, mem, cb);
  }

  {
    int n = NNODES * SDIM;
    k_copy_out<<<(n + 255) / 256, 256, 0, stream>>>(reprs, (float*)d_out, n);
  }
}
